// DEC_18073222381742
// MI455X (gfx1250) — matrix operand order repaired and checked by reading, not compiled, not run on hardware
//
#include <hip/hip_runtime.h>

// -------- CDNA5 WMMA types --------
typedef __attribute__((ext_vector_type(16))) __bf16 v16bf;
typedef __attribute__((ext_vector_type(8)))  float  v8f;
typedef __attribute__((ext_vector_type(4)))  int    v4i;

#define BDIM 1024
#define LSEQ 128
#define HDIM 256
#define EDIM 256
#define GDIM 768   // 3*H
#define DSH  10
#define APITCH 40  // LDS row pitch in bf16 (80B: 16B-aligned chunks, staggered banks)

// ---------------- async global->LDS staging (gfx1250) ----------------
#if defined(__has_builtin)
#  if __has_builtin(__builtin_amdgcn_global_load_async_to_lds_b128)
#    define HAVE_ASYNC_LDS 1
#  endif
#endif
#ifndef HAVE_ASYNC_LDS
#  define HAVE_ASYNC_LDS 0
#endif

typedef __attribute__((address_space(1))) v4i av1_v4i;
typedef __attribute__((address_space(3))) v4i av3_v4i;

__device__ __forceinline__ void async_copy16(const __bf16* g, __bf16* l) {
#if HAVE_ASYNC_LDS
  __builtin_amdgcn_global_load_async_to_lds_b128((av1_v4i*)g, (av3_v4i*)l, 0, 0);
#else
  *(int4*)l = *(const int4*)g;
#endif
}
__device__ __forceinline__ void async_wait_all() {
#if HAVE_ASYNC_LDS
  asm volatile("s_wait_asynccnt 0x0" ::: "memory");
#endif
}

__device__ __forceinline__ float sigmoid_f(float x) {
  return 1.0f / (1.0f + __expf(-x));
}
__device__ __forceinline__ float tanh_fast(float x) {
  float e = __expf(2.0f * x);
  return 1.0f - 2.0f / (e + 1.0f);   // correct limits at +/-inf
}

// ------------------------------------------------------------------
// C[M x N] = A[M x 256] * Bw[N x 256]^T (+ bias[N]), A/B bf16, C fp32.
// All operands have row pitch 256 bf16 (512B). M%64==0, N%128==0, K%32==0.
// Block: 256 thr = 8 waves; block tile 64x128; wave tile 32x32 (4 WMMA/chunk).
// Double-buffered LDS fed by async global->LDS B128 copies.
// ------------------------------------------------------------------
__global__ __launch_bounds__(256) void gemm_bf16w(
    const __bf16* __restrict__ A,
    const __bf16* __restrict__ Bw,
    const float* __restrict__ bias,
    float* __restrict__ C, int ldc, int K)
{
  __shared__ __bf16 As[2][64][APITCH];
  __shared__ __bf16 Bs[2][128][APITCH];

  const int tid  = threadIdx.x;
  const int wv   = tid >> 5;
  const int lane = tid & 31;
  const int hl   = lane >> 4;
  const int lr   = lane & 15;
  const long m0  = (long)blockIdx.x * 64;
  const long n0  = (long)blockIdx.y * 128;
  const int  wm  = (wv & 1) * 32;
  const int  wn  = (wv >> 1) * 32;

  v8f c00 = {}; v8f c01 = {}; v8f c10 = {}; v8f c11 = {};

  const int ar  = tid >> 2, aseg = tid & 3;        // A: 64x4 16B chunks
  const int br0 = tid >> 1, bseg0 = (tid & 1) * 2; // B: 128x4 chunks, 2/thread

  // stage(k0 -> buffer cb)
  auto stage = [&](int cb, int k0) {
    async_copy16(A + (m0 + ar) * 256 + k0 + aseg * 8, &As[cb][ar][aseg * 8]);
    async_copy16(Bw + (n0 + br0) * 256 + k0 + bseg0 * 8, &Bs[cb][br0][bseg0 * 8]);
    async_copy16(Bw + (n0 + br0) * 256 + k0 + (bseg0 + 1) * 8, &Bs[cb][br0][(bseg0 + 1) * 8]);
  };

  const int nIter = K >> 5;
  stage(0, 0);
  for (int it = 0; it < nIter; ++it) {
    const int cb = it & 1;
    async_wait_all();       // my async copies into buffer cb are done
    __syncthreads();        // everyone's copies done & prev compute retired
    if (it + 1 < nIter) stage(cb ^ 1, (it + 1) * 32);

    // A fragments (ISA 7.12.2, 16-bit A 16x32): VGPR i<4 -> K=2i(+1)+8*hl,
    // i>=4 -> K=16+2(i-4)(+1)+8*hl ; row M = lr.
    v16bf a0, a1, b0, b1;
    #pragma unroll
    for (int i = 0; i < 8; ++i) {
      int kk = ((i < 4) ? (2 * i) : (16 + 2 * (i - 4))) + 8 * hl;
      a0[2 * i]     = As[cb][wm + lr][kk];
      a0[2 * i + 1] = As[cb][wm + lr][kk + 1];
      a1[2 * i]     = As[cb][wm + 16 + lr][kk];
      a1[2 * i + 1] = As[cb][wm + 16 + lr][kk + 1];
    }
    // B fragment (32x16 KxN): element e -> K = (e<8 ? e : e+8) + 8*hl, N = lr; B[k][n]=Bw[n][k]
    #pragma unroll
    for (int e = 0; e < 16; ++e) {
      b0[e] = Bs[cb][wn + lr][((e < 8) ? e : (e + 8)) + 8 * hl];
      b1[e] = Bs[cb][wn + 16 + lr][((e < 8) ? e : (e + 8)) + 8 * hl];
    }

    c00 = __builtin_amdgcn_wmma_f32_16x16x32_bf16(false, a0, false, b0, (short)0, c00, false, false);
    c01 = __builtin_amdgcn_wmma_f32_16x16x32_bf16(false, a0, false, b1, (short)0, c01, false, false);
    c10 = __builtin_amdgcn_wmma_f32_16x16x32_bf16(false, a1, false, b0, (short)0, c10, false, false);
    c11 = __builtin_amdgcn_wmma_f32_16x16x32_bf16(false, a1, false, b1, (short)0, c11, false, false);
  }

  // C/D layout: VGPR i -> M = i + 8*hl ; N = lane lr
  const long col0 = n0 + wn + lr;
  const float bv0 = bias ? bias[col0] : 0.0f;
  const float bv1 = bias ? bias[col0 + 16] : 0.0f;
  #pragma unroll
  for (int i = 0; i < 8; ++i) {
    long r0 = m0 + wm + i + 8 * hl;
    C[r0 * (long)ldc + col0]             = c00[i] + bv0;
    C[r0 * (long)ldc + col0 + 16]        = c01[i] + bv1;
    C[(r0 + 16) * (long)ldc + col0]      = c10[i] + bv0;
    C[(r0 + 16) * (long)ldc + col0 + 16] = c11[i] + bv1;
  }
}

// ------------------------------------------------------------------
// fp32 -> bf16 matrix conversion; all matrices have 256 columns.
__global__ void cvt_bf16(const float* __restrict__ src, int ld,
                         __bf16* __restrict__ dst, long n)
{
  long i = (long)blockIdx.x * 256 + threadIdx.x;
  if (i >= n) return;
  long r = i >> 8, c = i & 255;
  dst[i] = (__bf16)src[r * (long)ld + c];
}

// ------------------------------------------------------------------
// GRU layer-0 gating. gi = x(B,3) @ w_ih^T inlined (K=3).
// gh buffer already contains h@w_hh^T + b_hh (from GEMM bias).
__global__ __launch_bounds__(256) void gru_l0_ew(
    const float* __restrict__ recv, int t,
    const float* __restrict__ w_ih, const float* __restrict__ b_ih,
    const float* __restrict__ gh,
    const float* __restrict__ hcur, float* __restrict__ hnext,
    __bf16* __restrict__ hnext_bf)
{
  int idx = blockIdx.x * 256 + threadIdx.x;   // B*H threads
  int b = idx >> 8, j = idx & 255;
  const float* xp = recv + ((size_t)b * LSEQ + t) * 3;
  float x0 = xp[0], x1 = xp[1], x2 = xp[2];
  const float* wr = w_ih + (size_t)j * 3;
  const float* wz = w_ih + (size_t)(j + 256) * 3;
  const float* wn = w_ih + (size_t)(j + 512) * 3;
  float gir = fmaf(wr[0], x0, fmaf(wr[1], x1, fmaf(wr[2], x2, b_ih[j])));
  float giz = fmaf(wz[0], x0, fmaf(wz[1], x1, fmaf(wz[2], x2, b_ih[j + 256])));
  float gin = fmaf(wn[0], x0, fmaf(wn[1], x1, fmaf(wn[2], x2, b_ih[j + 512])));
  const float* ghb = gh + (size_t)b * GDIM;
  float r = sigmoid_f(gir + ghb[j]);
  float z = sigmoid_f(giz + ghb[j + 256]);
  float n = tanh_fast(gin + r * ghb[j + 512]);
  float h = hcur[(size_t)b * HDIM + j];
  float hn = (1.0f - z) * n + z * h;
  hnext[(size_t)b * HDIM + j] = hn;
  hnext_bf[(size_t)b * HDIM + j] = (__bf16)hn;
}

// ------------------------------------------------------------------
// GRU layer-1 gating + reduction of o . w_o into s[b,t]. Block per b.
__global__ __launch_bounds__(256) void gru_l1_ew(
    const float* __restrict__ gi, const float* __restrict__ gh,
    const float* __restrict__ hcur, float* __restrict__ hnext,
    __bf16* __restrict__ hnext_bf,
    const float* __restrict__ w_o, float* __restrict__ s, int t)
{
  int b = blockIdx.x, j = threadIdx.x;
  const float* gib = gi + (size_t)b * GDIM;
  const float* ghb = gh + (size_t)b * GDIM;
  float r = sigmoid_f(gib[j] + ghb[j]);
  float z = sigmoid_f(gib[j + 256] + ghb[j + 256]);
  float n = tanh_fast(gib[j + 512] + r * ghb[j + 512]);
  float h = hcur[(size_t)b * HDIM + j];
  float hn = (1.0f - z) * n + z * h;
  hnext[(size_t)b * HDIM + j] = hn;
  hnext_bf[(size_t)b * HDIM + j] = (__bf16)hn;

  __shared__ float dr[256];
  dr[j] = hn * w_o[j];
  __syncthreads();
  for (int off = 128; off >= 1; off >>= 1) {
    if (j < off) dr[j] += dr[j + off];
    __syncthreads();
  }
  if (j == 0) s[(size_t)b * LSEQ + t] = dr[0];   // first writer this step
}

// ------------------------------------------------------------------
// Attention step: score/softmax/context + c . w_c into s[b,t].
// Block (8 waves) per batch row; each wave owns 16 source positions.
__global__ __launch_bounds__(256) void attn_kernel(
    const float* __restrict__ q,
    const float* __restrict__ attn_enc,
    const float* __restrict__ enc,
    const float* __restrict__ v,
    const float* __restrict__ w_c,
    float* __restrict__ s, int t)
{
  const int b = blockIdx.x;
  const int tid = threadIdx.x;
  const int wv = tid >> 5, lane = tid & 31;

  __shared__ float qs[HDIM];
  __shared__ float vs[HDIM];
  __shared__ float sc[LSEQ];
  __shared__ float red[LSEQ];
  __shared__ float dr[256];

  qs[tid] = q[(size_t)b * HDIM + tid];
  vs[tid] = v[tid];
  __syncthreads();

  const float* aeb = attn_enc + (size_t)b * LSEQ * HDIM;
  #pragma unroll 1
  for (int ii = 0; ii < 16; ++ii) {
    int si = wv * 16 + ii;
    const float* row = aeb + (size_t)si * HDIM;
    float p = 0.0f;
    for (int h = lane; h < HDIM; h += 32)
      p += vs[h] * tanh_fast(qs[h] + row[h]);
    #pragma unroll
    for (int off = 16; off >= 1; off >>= 1)
      p += __shfl_down(p, off, 32);
    if (lane == 0) sc[si] = p;
  }
  __syncthreads();

  if (tid < LSEQ) red[tid] = sc[tid];
  __syncthreads();
  for (int off = 64; off >= 1; off >>= 1) {
    if (tid < off) red[tid] = fmaxf(red[tid], red[tid + off]);
    __syncthreads();
  }
  float mx = red[0];
  __syncthreads();
  if (tid < LSEQ) {
    float e = __expf(sc[tid] - mx);
    sc[tid] = e;
    red[tid] = e;
  }
  __syncthreads();
  for (int off = 64; off >= 1; off >>= 1) {
    if (tid < off) red[tid] += red[tid + off];
    __syncthreads();
  }
  float inv = 1.0f / red[0];
  __syncthreads();

  const float* eb = enc + (size_t)b * LSEQ * EDIM;
  float c = 0.0f;
  for (int si = 0; si < LSEQ; ++si)
    c += sc[si] * eb[(size_t)si * EDIM + tid];
  c *= inv;

  dr[tid] = c * w_c[tid];
  __syncthreads();
  for (int off = 128; off >= 1; off >>= 1) {
    if (tid < off) dr[tid] += dr[tid + off];
    __syncthreads();
  }
  if (tid == 0) s[(size_t)b * LSEQ + t] += dr[0];
}

// ------------------------------------------------------------------
__global__ void init_h(const float* __restrict__ h0,
                       float* h00, float* h01, float* h10, float* h11,
                       __bf16* g00, __bf16* g01, __bf16* g10, __bf16* g11)
{
  int idx = blockIdx.x * 256 + threadIdx.x;   // B*H
  float l0 = h0[idx];
  float l1 = h0[BDIM * HDIM + idx];
  h00[idx] = l0; h01[idx] = l1;
  h10[idx] = l0; h11[idx] = l1;
  __bf16 b0 = (__bf16)l0, b1 = (__bf16)l1;
  g00[idx] = b0; g01[idx] = b1;
  g10[idx] = b0; g11[idx] = b1;
}

__global__ void final_kernel(const float* __restrict__ s1,
                             const float* __restrict__ s2,
                             const float* __restrict__ ob,
                             float* __restrict__ out)
{
  int idx = blockIdx.x * 256 + threadIdx.x;   // B*L
  int b = idx >> 7, t = idx & 127;
  int t2 = (t >= LSEQ - DSH - 1) ? (LSEQ - 1) : (t + DSH);
  out[idx] = sigmoid_f(s1[idx] + s2[(size_t)b * LSEQ + t2] + ob[0]);
}

// ------------------------------------------------------------------
extern "C" void kernel_launch(void* const* d_in, const int* in_sizes, int n_in,
                              void* d_out, int out_size, void* d_ws, size_t ws_size,
                              hipStream_t stream) {
  (void)in_sizes; (void)n_in; (void)out_size; (void)ws_size;
  const float* recv = (const float*)d_in[0];
  const float* enc  = (const float*)d_in[1];
  const float* h0   = (const float*)d_in[2];
  const float* w_ih0[2] = {(const float*)d_in[3],  (const float*)d_in[11]};
  const float* w_hh0[2] = {(const float*)d_in[4],  (const float*)d_in[12]};
  const float* b_ih0[2] = {(const float*)d_in[5],  (const float*)d_in[13]};
  const float* b_hh0[2] = {(const float*)d_in[6],  (const float*)d_in[14]};
  const float* w_ih1[2] = {(const float*)d_in[7],  (const float*)d_in[15]};
  const float* w_hh1[2] = {(const float*)d_in[8],  (const float*)d_in[16]};
  const float* b_ih1[2] = {(const float*)d_in[9],  (const float*)d_in[17]};
  const float* b_hh1[2] = {(const float*)d_in[10], (const float*)d_in[18]};
  const float* attn_W = (const float*)d_in[19];   // (H, H+E): [:, :H]=W_s  [:, H:]=W_e
  const float* v_W    = (const float*)d_in[20];
  const float* out_W  = (const float*)d_in[21];   // [w_o1 | w_c1 | w_o2 | w_c2]
  const float* out_b  = (const float*)d_in[22];
  float* out = (float*)d_out;

  // ---- workspace carve-up (fp32 region first, then bf16; all 16B aligned)
  char* wsb = (char*)d_ws;
  auto allocF = [&](size_t n) { float* p = (float*)wsb; wsb += n * sizeof(float); return p; };
  auto allocB = [&](size_t n) { __bf16* p = (__bf16*)wsb; wsb += n * sizeof(__bf16); return p; };

  float* attn_enc = allocF((size_t)BDIM * LSEQ * HDIM);
  float* hbuf[2][2][2];                               // [stack][layer][buf]
  for (int r = 0; r < 2; ++r)
    for (int l = 0; l < 2; ++l)
      for (int p = 0; p < 2; ++p) hbuf[r][l][p] = allocF((size_t)BDIM * HDIM);
  float* bufGH = allocF((size_t)BDIM * GDIM);
  float* bufGI = allocF((size_t)BDIM * GDIM);
  float* bufQ  = allocF((size_t)BDIM * HDIM);
  float* sbuf[2] = {allocF((size_t)BDIM * LSEQ), allocF((size_t)BDIM * LSEQ)};

  __bf16* enc_bf = allocB((size_t)BDIM * LSEQ * EDIM);
  __bf16* hbf[2][2][2];
  for (int r = 0; r < 2; ++r)
    for (int l = 0; l < 2; ++l)
      for (int p = 0; p < 2; ++p) hbf[r][l][p] = allocB((size_t)BDIM * HDIM);
  __bf16* wb_hh0[2], *wb_ih1[2], *wb_hh1[2];
  for (int r = 0; r < 2; ++r) {
    wb_hh0[r] = allocB((size_t)GDIM * HDIM);
    wb_ih1[r] = allocB((size_t)GDIM * HDIM);
    wb_hh1[r] = allocB((size_t)GDIM * HDIM);
  }
  __bf16* Ws_bf = allocB((size_t)HDIM * HDIM);
  __bf16* We_bf = allocB((size_t)HDIM * EDIM);

  const float* w_o[2] = {out_W + 0,   out_W + 512};
  const float* w_c[2] = {out_W + 256, out_W + 768};

  // ---- prologue: bf16 conversions, attn_enc GEMM, hidden init
  {
    long nEnc = (long)BDIM * LSEQ * EDIM;
    cvt_bf16<<<(unsigned)((nEnc + 255) / 256), 256, 0, stream>>>(enc, EDIM, enc_bf, nEnc);
    long nW = (long)GDIM * HDIM;
    for (int r = 0; r < 2; ++r) {
      cvt_bf16<<<(unsigned)((nW + 255) / 256), 256, 0, stream>>>(w_hh0[r], HDIM, wb_hh0[r], nW);
      cvt_bf16<<<(unsigned)((nW + 255) / 256), 256, 0, stream>>>(w_ih1[r], HDIM, wb_ih1[r], nW);
      cvt_bf16<<<(unsigned)((nW + 255) / 256), 256, 0, stream>>>(w_hh1[r], HDIM, wb_hh1[r], nW);
    }
    long nS = (long)HDIM * HDIM;
    cvt_bf16<<<(unsigned)((nS + 255) / 256), 256, 0, stream>>>(attn_W, HDIM + EDIM, Ws_bf, nS);
    cvt_bf16<<<(unsigned)((nS + 255) / 256), 256, 0, stream>>>(attn_W + HDIM, HDIM + EDIM, We_bf, nS);
  }
  // attn_enc = enc @ W_e^T  (M=B*L, N=H, K=E)
  gemm_bf16w<<<dim3((BDIM * LSEQ) / 64, HDIM / 128), 256, 0, stream>>>(
      enc_bf, We_bf, nullptr, attn_enc, HDIM, EDIM);
  init_h<<<(BDIM * HDIM) / 256, 256, 0, stream>>>(
      h0, hbuf[0][0][0], hbuf[0][1][0], hbuf[1][0][0], hbuf[1][1][0],
      hbf[0][0][0], hbf[0][1][0], hbf[1][0][0], hbf[1][1][0]);

  for (int t = 0; t < LSEQ; ++t) {
    int cur = t & 1, nxt = cur ^ 1;
    for (int r = 0; r < 2; ++r) {
      // gh0 = h_l0 @ w_hh0^T + b_hh0
      gemm_bf16w<<<dim3(BDIM / 64, GDIM / 128), 256, 0, stream>>>(
          hbf[r][0][cur], wb_hh0[r], b_hh0[r], bufGH, GDIM, HDIM);
      gru_l0_ew<<<(BDIM * HDIM) / 256, 256, 0, stream>>>(
          recv, t, w_ih0[r], b_ih0[r], bufGH, hbuf[r][0][cur], hbuf[r][0][nxt], hbf[r][0][nxt]);
      // gi1 = h0n @ w_ih1^T + b_ih1 ; gh1 = h_l1 @ w_hh1^T + b_hh1
      gemm_bf16w<<<dim3(BDIM / 64, GDIM / 128), 256, 0, stream>>>(
          hbf[r][0][nxt], wb_ih1[r], b_ih1[r], bufGI, GDIM, HDIM);
      gemm_bf16w<<<dim3(BDIM / 64, GDIM / 128), 256, 0, stream>>>(
          hbf[r][1][cur], wb_hh1[r], b_hh1[r], bufGH, GDIM, HDIM);
      gru_l1_ew<<<BDIM, 256, 0, stream>>>(
          bufGI, bufGH, hbuf[r][1][cur], hbuf[r][1][nxt], hbf[r][1][nxt],
          w_o[r], sbuf[r], t);
      // attention uses PRE-update h_l1: q = h_l1 @ W_s^T
      gemm_bf16w<<<dim3(BDIM / 64, HDIM / 128), 256, 0, stream>>>(
          hbf[r][1][cur], Ws_bf, nullptr, bufQ, HDIM, HDIM);
      attn_kernel<<<BDIM, 256, 0, stream>>>(
          bufQ, attn_enc, enc, v_W, w_c[r], sbuf[r], t);
    }
  }

  final_kernel<<<(BDIM * LSEQ) / 256, 256, 0, stream>>>(sbuf[0], sbuf[1], out_b, out);
}
